// MultiHeadAtt_79517024518750
// MI455X (gfx1250) — matrix operand order repaired and checked by reading, not compiled, not run on hardware
//
#include <hip/hip_runtime.h>
#include <stdint.h>

// ---------------------------------------------------------------- types ----
typedef __attribute__((ext_vector_type(8)))  __bf16 v8bf;
typedef __attribute__((ext_vector_type(16))) __bf16 v16bf;
typedef __attribute__((ext_vector_type(8)))  float  v8f;
typedef __attribute__((ext_vector_type(4)))  unsigned int u32x4;
typedef __attribute__((ext_vector_type(8)))  int    i32x8;
typedef __attribute__((ext_vector_type(4)))  int    i32x4;

#if defined(__has_builtin)
#if __has_builtin(__builtin_amdgcn_tensor_load_to_lds) && __has_builtin(__builtin_amdgcn_s_wait_tensorcnt)
#define HAVE_TDM 1
#endif
#endif

#if __has_include(<hip/amd_detail/amd_gfx1250_TDM.h>)
#define TDM_6ARG 1
#endif

static constexpr int    BATCH = 32;
static constexpr int    SEQ   = 2048;
static constexpr int    EMB   = 1024;
static constexpr int    NH    = 16;
static constexpr int    DH    = 64;
static constexpr int    HD    = NH * DH;              // 1024
static constexpr size_t MROWS = (size_t)BATCH * SEQ;  // 65536
static constexpr int    DFLAT = SEQ * EMB;            // 2097152

__device__ __forceinline__ unsigned short f32_to_bf16_raw(float f) {
  union { float f; unsigned int u; } c; c.f = f;
  unsigned int u = c.u + 0x7FFFu + ((c.u >> 16) & 1u);  // RNE
  return (unsigned short)(u >> 16);
}
__device__ __forceinline__ float bf16_raw_to_f32(unsigned short h) {
  union { unsigned int u; float f; } c; c.u = ((unsigned int)h) << 16;
  return c.f;
}

#if defined(HAVE_TDM)
// ---------------------------------------------------------------- TDM -------
// Issue a TENSOR_LOAD_TO_LDS for a 2-D bf16 tile: tile_w x tile_h elements,
// row stride = stride_elems, from gptr into LDS byte offset lds_off.
// D# layout per CDNA5 ISA ch.8: group0 = {count/flags, lds_addr, gaddr,
// gaddr_hi|type=2}; group1 = {data_size, tensor dims, tile dims, stride}.
// tensor_dim0/1 set huge (2^30): every tile here is fully in-bounds, the
// dims only have to avoid clipping.
__device__ __forceinline__ void tdm_load_2d_bf16(unsigned lds_off,
                                                 const unsigned short* gptr,
                                                 int tile_w, int tile_h,
                                                 int stride_elems) {
  unsigned long long ga = (unsigned long long)(size_t)gptr;
  u32x4 g0;
  g0[0] = 1u;                                              // count=1, user D#
  g0[1] = lds_off;                                         // lds_addr (bytes)
  g0[2] = (unsigned)ga;                                    // global_addr[31:0]
  g0[3] = (unsigned)((ga >> 32) & 0x01FFFFFFu) | 0x80000000u; // [56:32]|type=2
  i32x8 g1;
  g1[0] = (int)(1u << 16);                                 // data_size=1 (2B)
  g1[1] = 0;                                               // tensor_dim0 lo16=0
  g1[2] = 0x4000;                                          // tensor_dim0=2^30
  g1[3] = (int)(0x4000u | ((unsigned)tile_w << 16));       // tensor_dim1=2^30, tile_dim0
  g1[4] = tile_h;                                          // tile_dim1 (tile_dim2=0)
  g1[5] = stride_elems;                                    // tensor_dim0_stride lo32
  g1[6] = 0;                                               // stride hi, dim1_stride lo
  g1[7] = 0;
  i32x4 z4 = (i32x4)0;
#if defined(TDM_6ARG)
  i32x8 z8 = (i32x8)0;
  __builtin_amdgcn_tensor_load_to_lds(g0, g1, z4, z4, z8, 0);
#else
  __builtin_amdgcn_tensor_load_to_lds(g0, g1, z4, z4, 0);
#endif
}
#endif  // HAVE_TDM

// ------------------------------------------------ weight convert+transpose --
// W [K][N] f32 row-major  ->  Wt [N][K] bf16 row-major
__global__ void convert_transpose(const float* __restrict__ W,
                                  unsigned short* __restrict__ Wt,
                                  int K, int N) {
  size_t idx = (size_t)blockIdx.x * 256 + threadIdx.x;
  if (idx >= (size_t)K * N) return;
  int k = (int)(idx / N);
  int n = (int)(idx % N);
  Wt[(size_t)n * K + k] = f32_to_bf16_raw(W[idx]);
}

// ---------------------------------------------------------- embed gather ----
__global__ __launch_bounds__(256)
void gather_embed(const int* __restrict__ x, const float* __restrict__ embed,
                  unsigned short* __restrict__ h) {
  int row = blockIdx.x;
  int e   = threadIdx.x * 4;
  size_t src = (size_t)x[row] * EMB + e;
  size_t dst = (size_t)row * EMB + e;
  float4 f = *(const float4*)&embed[src];
  unsigned short o0 = f32_to_bf16_raw(f.x);
  unsigned short o1 = f32_to_bf16_raw(f.y);
  unsigned short o2 = f32_to_bf16_raw(f.z);
  unsigned short o3 = f32_to_bf16_raw(f.w);
  uint2 pk;
  pk.x = (unsigned)o0 | ((unsigned)o1 << 16);
  pk.y = (unsigned)o2 | ((unsigned)o3 << 16);
  *(uint2*)&h[dst] = pk;
}

// ------------------------------------------------------------ WMMA GEMM -----
// C[M][N](bf16) = A[M][K](bf16) @ Bt[N][K](bf16)^T  (+ resid[M][N] if given)
// 256 threads = 8 waves, 128x128 block tile, K-step 32, double-buffered LDS.
// TDM path: wave 0 drives the Tensor Data Mover (TENSORcnt) while the tile
// for the previous K-step is consumed by WMMA -> DMA/compute overlap.
__global__ __launch_bounds__(256)
void gemm_bf16_wmma(const unsigned short* __restrict__ A,
                    const unsigned short* __restrict__ Bt,
                    const unsigned short* resid,
                    unsigned short* C,
                    int N, int K) {
  __shared__ unsigned short As[2][128 * 32];
  __shared__ unsigned short Bs[2][128 * 32];

  const int tid  = threadIdx.x;
  const int wave = tid >> 5;
  const int lane = tid & 31;
  const int half = lane >> 4;   // 0: lanes 0-15, 1: lanes 16-31
  const int l16  = lane & 15;
  const size_t rowBase = (size_t)blockIdx.y * 128;
  const size_t colBase = (size_t)blockIdx.x * 128;

  v8f acc[8] = {};

#if defined(HAVE_TDM)
  if (wave == 0) {  // prologue: DMA first K-tile pair into buffer 0
    tdm_load_2d_bf16((unsigned)(size_t)&As[0][0], &A [rowBase * K], 32, 128, K);
    tdm_load_2d_bf16((unsigned)(size_t)&Bs[0][0], &Bt[colBase * K], 32, 128, K);
  }
#endif

  for (int k0 = 0; k0 < K; k0 += 32) {
    const int buf = (k0 >> 5) & 1;
#if defined(HAVE_TDM)
    if (wave == 0) __builtin_amdgcn_s_wait_tensorcnt(0);  // current tile landed
    __syncthreads();
    if (wave == 0 && (k0 + 32 < K)) {  // prefetch next K-tile into other buffer
      tdm_load_2d_bf16((unsigned)(size_t)&As[buf ^ 1][0],
                       &A [rowBase * K + k0 + 32], 32, 128, K);
      tdm_load_2d_bf16((unsigned)(size_t)&Bs[buf ^ 1][0],
                       &Bt[colBase * K + k0 + 32], 32, 128, K);
    }
#else
    // fallback: synchronous cooperative staging (16B chunks)
#pragma unroll
    for (int s = 0; s < 2; ++s) {
      int c  = tid + s * 256;
      int r  = c >> 2;
      int kc = (c & 3) * 8;
      *(uint4*)&As[buf][r * 32 + kc] = *(const uint4*)&A [(rowBase + r) * K + k0 + kc];
      *(uint4*)&Bs[buf][r * 32 + kc] = *(const uint4*)&Bt[(colBase + r) * K + k0 + kc];
    }
    if (k0 + 32 < K) {
      __builtin_prefetch(&A [(rowBase + (tid >> 1)) * K + k0 + 32], 0, 1);
      __builtin_prefetch(&Bt[(colBase + (tid >> 1)) * K + k0 + 32], 0, 1);
    }
    __syncthreads();
#endif
    const unsigned short* Ab = &As[buf][0];
    const unsigned short* Bb = &Bs[buf][0];

    // A fragment: 16x32 (VGPR0-3: K=half*16+e, VGPR4-7: K=half*16+8+e)
    const v8bf alo = *(const v8bf*)&Ab[(wave * 16 + l16) * 32 + half * 16];
    const v8bf ahi = *(const v8bf*)&Ab[(wave * 16 + l16) * 32 + half * 16 + 8];
    v16bf afrag = __builtin_shufflevector(alo, ahi,
        0,1,2,3,4,5,6,7,8,9,10,11,12,13,14,15);

    // preload all 8 B fragments, then a back-to-back WMMA chain
    v16bf bfrag[8];
#pragma unroll
    for (int sub = 0; sub < 8; ++sub) {
      const v8bf blo = *(const v8bf*)&Bb[(sub * 16 + l16) * 32 + half * 16];
      const v8bf bhi = *(const v8bf*)&Bb[(sub * 16 + l16) * 32 + half * 16 + 8];
      bfrag[sub] = __builtin_shufflevector(blo, bhi,
          0,1,2,3,4,5,6,7,8,9,10,11,12,13,14,15);
    }
#pragma unroll
    for (int sub = 0; sub < 8; ++sub) {
      acc[sub] = __builtin_amdgcn_wmma_f32_16x16x32_bf16(
          false, afrag, false, bfrag[sub], (short)0, acc[sub], false, false);
    }
    __syncthreads();  // all reads of `buf` done before it is refilled
  }

  // epilogue: D layout VGPR r -> row r+8*half, col lane%16
#pragma unroll
  for (int sub = 0; sub < 8; ++sub) {
#pragma unroll
    for (int r = 0; r < 8; ++r) {
      size_t grow = rowBase + wave * 16 + r + half * 8;
      size_t gcol = colBase + sub * 16 + l16;
      float v = acc[sub][r];
      if (resid) v += bf16_raw_to_f32(resid[grow * N + gcol]);
      C[grow * N + gcol] = f32_to_bf16_raw(v);
    }
  }
}

// ------------------------------------------------------- batch attention ----
// Per (l, head) site: Q,K,V are [32 x 64] over the *batch* axis.
// One wave per site; lane i = batch index. att may alias q.
__global__ __launch_bounds__(32)
void attention_kernel(const unsigned short* q,
                      const unsigned short* kv,
                      unsigned short* att) {
  __shared__ float Qs[32][64];
  __shared__ float Ks[32][64];
  __shared__ float Vs[32][64];
  __shared__ float Ss[32][32];

  const int site = blockIdx.x;
  const int l = site >> 4;
  const int n = site & 15;
  const int i = threadIdx.x;

  const size_t rowq  = ((size_t)i * SEQ + l) * HD + n * DH;
  const size_t rowkv = ((size_t)i * SEQ + l) * (2 * HD);
  for (int d = 0; d < DH; ++d) {
    Qs[i][d] = bf16_raw_to_f32(q [rowq + d]);
    Ks[i][d] = bf16_raw_to_f32(kv[rowkv + n * DH + d]);
    Vs[i][d] = bf16_raw_to_f32(kv[rowkv + HD + n * DH + d]);
  }
  __syncthreads();

  const float scale = 0.125f;  // 1/sqrt(64)
  float mx = -3.0e38f;
  for (int j = 0; j < 32; ++j) {
    float s = 0.0f;
    for (int d = 0; d < DH; ++d) s += Qs[i][d] * Ks[j][d];
    s *= scale;
    Ss[i][j] = s;
    mx = fmaxf(mx, s);
  }
  float sum = 0.0f;
  for (int j = 0; j < 32; ++j) {
    float e = __expf(Ss[i][j] - mx);
    Ss[i][j] = e;
    sum += e;
  }
  const float inv = 1.0f / sum;
  for (int d = 0; d < DH; ++d) {
    float a = 0.0f;
    for (int j = 0; j < 32; ++j) a += Ss[i][j] * Vs[j][d];
    att[rowq + d] = f32_to_bf16_raw(a * inv);
  }
}

// --------------------------------------------- classifier partial reduce ----
__global__ __launch_bounds__(256)
void logits_partial(const unsigned short* __restrict__ out,
                    const float* __restrict__ Wf,
                    float* __restrict__ partials) {
  const int b = blockIdx.y;
  const int chunk = blockIdx.x;
  const int t = threadIdx.x;
  const int per = DFLAT / 256;        // 8192
  const int start = chunk * per;
  const size_t base = (size_t)b * DFLAT;

  float a0 = 0.0f, a1 = 0.0f;
  for (int idx = start + t; idx < start + per; idx += 256) {
    float o = bf16_raw_to_f32(out[base + idx]);
    a0 += o * Wf[(size_t)idx * 2 + 0];
    a1 += o * Wf[(size_t)idx * 2 + 1];
  }
  __shared__ float r0[256], r1[256];
  r0[t] = a0; r1[t] = a1;
  __syncthreads();
  for (int s = 128; s > 0; s >>= 1) {
    if (t < s) { r0[t] += r0[t + s]; r1[t] += r1[t + s]; }
    __syncthreads();
  }
  if (t == 0) {
    partials[((size_t)b * 256 + chunk) * 2 + 0] = r0[0];
    partials[((size_t)b * 256 + chunk) * 2 + 1] = r1[0];
  }
}

// deterministic final sum: logits[b][c] = bf[c] + sum_j partials[b][j][c]
__global__ __launch_bounds__(64)
void finalize_logits(const float* __restrict__ partials,
                     const float* __restrict__ bfv,
                     float* __restrict__ logits) {
  int t = threadIdx.x;
  if (t >= BATCH * 2) return;
  int b = t >> 1, c = t & 1;
  float s = bfv[c];
  for (int j = 0; j < 256; ++j) s += partials[((size_t)b * 256 + j) * 2 + c];
  logits[b * 2 + c] = s;
}

// ------------------------------------------------------- CE loss (C = 2) ----
__global__ __launch_bounds__(32)
void loss_kernel(const float* __restrict__ logits, const int* __restrict__ y,
                 float* __restrict__ out) {
  __shared__ float red[32];
  int b = threadIdx.x;
  float l0 = logits[b * 2 + 0];
  float l1 = logits[b * 2 + 1];
  float m  = fmaxf(l0, l1);
  float lse = m + __logf(__expf(l0 - m) + __expf(l1 - m));
  float ly  = (y[b] == 0) ? l0 : l1;
  red[b] = -(ly - lse) * (1.0f / BATCH);
  __syncthreads();
  for (int s = 16; s > 0; s >>= 1) {
    if (b < s) red[b] += red[b + s];
    __syncthreads();
  }
  if (b == 0) out[0] = red[0];
}

// ------------------------------------------------------------- launcher -----
extern "C" void kernel_launch(void* const* d_in, const int* in_sizes, int n_in,
                              void* d_out, int out_size, void* d_ws, size_t ws_size,
                              hipStream_t stream) {
  (void)in_sizes; (void)n_in; (void)out_size; (void)ws_size;

  const int*   x     = (const int*)  d_in[0];
  const int*   y     = (const int*)  d_in[1];
  const float* embed = (const float*)d_in[2];
  const float* Wq    = (const float*)d_in[3];
  const float* Wkv   = (const float*)d_in[4];
  const float* Wo    = (const float*)d_in[5];
  const float* Wf    = (const float*)d_in[6];
  const float* bfv   = (const float*)d_in[7];
  float*       loss  = (float*)d_out;

  char* ws = (char*)d_ws;
  size_t off = 0;
  auto carve = [&](size_t bytes) -> void* {
    void* p = ws + off;
    off = (off + bytes + 255) & ~(size_t)255;
    return p;
  };
  unsigned short* h    = (unsigned short*)carve(MROWS * EMB * 2);
  unsigned short* q    = (unsigned short*)carve(MROWS * HD * 2);      // att in-place
  unsigned short* kv   = (unsigned short*)carve(MROWS * 2 * HD * 2);
  unsigned short* WqT  = (unsigned short*)carve((size_t)HD * EMB * 2);
  unsigned short* WkvT = (unsigned short*)carve((size_t)2 * HD * EMB * 2);
  unsigned short* WoT  = (unsigned short*)carve((size_t)EMB * HD * 2);
  float*          part = (float*)carve((size_t)BATCH * 256 * 2 * 4);
  float*          lgts = (float*)carve(64 * 4);

  convert_transpose<<<(EMB * HD + 255) / 256, 256, 0, stream>>>(Wq, WqT, EMB, HD);
  convert_transpose<<<(EMB * 2 * HD + 255) / 256, 256, 0, stream>>>(Wkv, WkvT, EMB, 2 * HD);
  convert_transpose<<<(HD * EMB + 255) / 256, 256, 0, stream>>>(Wo, WoT, HD, EMB);

  gather_embed<<<(unsigned)MROWS, 256, 0, stream>>>(x, embed, h);

  gemm_bf16_wmma<<<dim3(HD / 128, (unsigned)(MROWS / 128)), 256, 0, stream>>>(
      h, WqT, nullptr, q, HD, EMB);
  gemm_bf16_wmma<<<dim3(2 * HD / 128, (unsigned)(MROWS / 128)), 256, 0, stream>>>(
      h, WkvT, nullptr, kv, 2 * HD, EMB);

  attention_kernel<<<SEQ * NH, 32, 0, stream>>>(q, kv, q);

  gemm_bf16_wmma<<<dim3(EMB / 128, (unsigned)(MROWS / 128)), 256, 0, stream>>>(
      q, WoT, h, h, EMB, HD);

  logits_partial<<<dim3(256, BATCH), 256, 0, stream>>>(h, Wf, part);
  finalize_logits<<<1, 64, 0, stream>>>(part, bfv, lgts);
  loss_kernel<<<1, 32, 0, stream>>>(lgts, y, loss);
}
